// RoformerAttention_10084583211363
// MI455X (gfx1250) — matrix operand order repaired and checked by reading, not compiled, not run on hardware
//
#include <hip/hip_runtime.h>
#include <hip/hip_bf16.h>
#include <math.h>

#define B_  2
#define S_  2048
#define D_  1024
#define H_  16
#define HD_ 64
#define M_  (B_*S_)   // 4096 rows of the flattened activations

typedef _Float16 half_t;
typedef _Float16 v16h __attribute__((ext_vector_type(16)));
typedef _Float16 v8h  __attribute__((ext_vector_type(8)));
typedef _Float16 v2h  __attribute__((ext_vector_type(2)));
typedef __fp16   v2hp __attribute__((ext_vector_type(2)));   // cvt_pkrtz result type
typedef float    v8f  __attribute__((ext_vector_type(8)));
typedef unsigned int u32x4 __attribute__((ext_vector_type(4)));
typedef int          i32x4 __attribute__((ext_vector_type(4)));
typedef int          i32x8 __attribute__((ext_vector_type(8)));

// ---- CDNA5 feature detection (TDM builtin arity differs across toolchains) --
#if defined(__has_builtin)
# if __has_builtin(__builtin_amdgcn_tensor_load_to_lds)
#  define TDM_OK 1
# endif
# if __has_builtin(__builtin_amdgcn_s_wait_tensorcnt)
#  define HAVE_WAIT_TCNT 1
# endif
#endif
#if __has_include(<hip/amd_detail/amd_gfx1250_TDM.h>)
# define TDM_SIX 1   // therock headers -> 6-arg builtin
#endif

static __device__ __forceinline__ void wait_tensorcnt0() {
#if defined(HAVE_WAIT_TCNT)
  __builtin_amdgcn_s_wait_tensorcnt(0);
#else
  asm volatile("s_wait_tensorcnt 0x0" ::: "memory");
#endif
}

static __device__ __forceinline__ v16h cat16(v8h lo, v8h hi) {
  return __builtin_shufflevector(lo, hi, 0,1,2,3,4,5,6,7,8,9,10,11,12,13,14,15);
}
static __device__ __forceinline__ v8f wmma16(v16h a, v16h b, v8f c) {
  // D = A(16x32 f16) * B(32x16 f16) + C(16x16 f32)
  return __builtin_amdgcn_wmma_f32_16x16x32_f16(false, a, false, b, (short)0, c, false, false);
}
static __device__ __forceinline__ v8f zero8() {
  v8f z;
#pragma unroll
  for (int i = 0; i < 8; ++i) z[i] = 0.0f;
  return z;
}
// pack two f32 into f16x2 (v_cvt_pk_rtz_f16_f32) and store as 4 bytes
static __device__ __forceinline__ void store_pk(half_t* dst, float a, float b) {
  v2hp pk = __builtin_amdgcn_cvt_pkrtz(a, b);
  *(v2hp*)dst = pk;
}

// -----------------------------------------------------------------------------
// Kernel 1: fused QKV projection (+bias, +RoPE for Q/K) with f16 WMMA.
// grid = (D/64, M/128, 3), block = 256 (8 waves). Wave computes 32x32 output.
// Writes Q/K/V as f16 in [B, H, S, HD] layout.
// -----------------------------------------------------------------------------
__global__ __launch_bounds__(256)
void qkv_rope_kernel(const float* __restrict__ x,
                     const float* __restrict__ Wq, const float* __restrict__ bq,
                     const float* __restrict__ Wk, const float* __restrict__ bk,
                     const float* __restrict__ Wv, const float* __restrict__ bv,
                     half_t* __restrict__ Qh, half_t* __restrict__ Kh,
                     half_t* __restrict__ Vh)
{
  __shared__ __align__(16) half_t As[128 * 32];   // A tile [row][k]
  __shared__ __align__(16) half_t Bs[64 * 32];    // B tile transposed [n][k]

  const int mode = blockIdx.z;                    // 0=Q 1=K 2=V
  const float* W    = (mode == 0) ? Wq : (mode == 1) ? Wk : Wv;
  const float* bias = (mode == 0) ? bq : (mode == 1) ? bk : bv;
  half_t* out       = (mode == 0) ? Qh : (mode == 1) ? Kh : Vh;

  const int Nbase = blockIdx.x * 64;
  const int Mbase = blockIdx.y * 128;
  const int t = threadIdx.x;
  const int w = t >> 5, lane = t & 31;
  const int hi = lane >> 4, nl = lane & 15;
  const int wm = w >> 1, wn = w & 1;              // 4x2 wave grid

  v8f acc[2][2];
#pragma unroll
  for (int mi = 0; mi < 2; ++mi)
#pragma unroll
    for (int ni = 0; ni < 2; ++ni) acc[mi][ni] = zero8();

  for (int kb = 0; kb < D_; kb += 32) {
    // stage A (f32 -> packed f16): 128x32, 8 pairs/thread
#pragma unroll
    for (int i = 0; i < 8; ++i) {
      int p = t + 256 * i;                 // pair index
      int row = p >> 4, col = (p & 15) * 2;
      const float* xp = &x[(size_t)(Mbase + row) * D_ + kb + col];
      store_pk(As + row * 32 + col, xp[0], xp[1]);
    }
    // stage B transposed (f32 -> packed f16): Bs[n][kk] = W[kb+kk][Nbase+n]
#pragma unroll
    for (int i = 0; i < 4; ++i) {
      int p = t + 256 * i;
      int n = p & 63, kk = (p >> 6) * 2;
      const float* wp = &W[(size_t)(kb + kk) * D_ + Nbase + n];
      store_pk(Bs + n * 32 + kk, wp[0], wp[D_]);
    }
    __syncthreads();

    if (kb + 32 < D_) {   // global_prefetch_b8 for next tiles
      __builtin_prefetch(&x[(size_t)(Mbase + (t >> 1)) * D_ + kb + 32], 0, 0);
      __builtin_prefetch(&W[(size_t)(kb + 32 + (t >> 6)) * D_ + Nbase + (t & 63)], 0, 0);
    }

    v16h a[2], b[2];
#pragma unroll
    for (int mi = 0; mi < 2; ++mi) {
      const half_t* ap = &As[(wm * 32 + mi * 16 + nl) * 32];
      a[mi] = cat16(*(const v8h*)(ap + 8 * hi), *(const v8h*)(ap + 16 + 8 * hi));
    }
#pragma unroll
    for (int ni = 0; ni < 2; ++ni) {
      const half_t* bp = &Bs[(wn * 32 + ni * 16 + nl) * 32];
      b[ni] = cat16(*(const v8h*)(bp + 8 * hi), *(const v8h*)(bp + 16 + 8 * hi));
    }
#pragma unroll
    for (int mi = 0; mi < 2; ++mi)
#pragma unroll
      for (int ni = 0; ni < 2; ++ni)
        acc[mi][ni] = wmma16(a[mi], b[ni], acc[mi][ni]);
    __syncthreads();
  }

  // epilogue: bias (+RoPE for Q/K), scatter to [B,H,S,HD] f16
#pragma unroll
  for (int ni = 0; ni < 2; ++ni) {
    const int n = Nbase + wn * 32 + ni * 16 + nl;
    const float bval = bias[n];
    const int dd = n & 63, head = n >> 6;
    const float invf = __powf(10000.0f, -(float)(dd & 31) * (1.0f / 32.0f));
    const float sgn = (n & 1) ? 1.0f : -1.0f;
#pragma unroll
    for (int mi = 0; mi < 2; ++mi) {
#pragma unroll
      for (int r = 0; r < 8; ++r) {
        const int m = Mbase + wm * 32 + mi * 16 + r + 8 * hi;
        float v = acc[mi][ni][r] + bval;
        if (mode < 2) {
          // partner element (n^1) lives in the adjacent lane in C-layout
          float pv = __shfl_xor(v, 1, 32);
          const int pos = m & (S_ - 1);
          const float ang = (float)pos * invf;
          v = v * __cosf(ang) + sgn * pv * __sinf(ang);
        }
        const int bb = m >> 11, s = m & (S_ - 1);
        out[(((size_t)(bb * H_ + head)) * S_ + s) * HD_ + dd] = (half_t)v;
      }
    }
  }
}

// -----------------------------------------------------------------------------
// Kernel 2: flash attention. grid = (S/128, B*H), block = 256 (8 waves).
// K tile staged with the Tensor Data Mover (TDM) when available; V tile staged
// transposed with packed stores. Online softmax; 8 WMMAs per key-block.
// -----------------------------------------------------------------------------
__global__ __launch_bounds__(256)
void attention_kernel(const half_t* __restrict__ Qh, const half_t* __restrict__ Kh,
                      const half_t* __restrict__ Vh, const float* __restrict__ cdr_bias,
                      const float* __restrict__ cdr_weight, half_t* __restrict__ attnh)
{
  __shared__ __align__(16) half_t Ks[32 * 64];    // K block [key][hd]   (== B^T for scores)
  __shared__ __align__(16) half_t VTs[64 * 32];   // V block transposed [hd][key]
  __shared__ __align__(16) half_t Ps[8][16 * 32]; // per-wave P staging [row][key]

  const int bh = blockIdx.y;
  const int b = bh >> 4, h = bh & 15;
  const int qbase = blockIdx.x * 128;
  const int t = threadIdx.x, w = t >> 5, lane = t & 31;
  const int hi = lane >> 4, nl = lane & 15;
  const float cw = cdr_weight[0];

  // Q A-fragments for this wave's 16 rows (held in registers for whole loop)
  const int arow = qbase + w * 16 + nl;
  const half_t* Qp = Qh + ((size_t)bh * S_ + arow) * HD_;
  v16h aq[2];
#pragma unroll
  for (int k2 = 0; k2 < 2; ++k2)
    aq[k2] = cat16(*(const v8h*)(Qp + 32 * k2 + 8 * hi),
                   *(const v8h*)(Qp + 32 * k2 + 16 + 8 * hi));

  float rm[8], rl[8];
  v8f o[4];
#pragma unroll
  for (int r = 0; r < 8; ++r) { rm[r] = -1e30f; rl[r] = 0.0f; }
#pragma unroll
  for (int nf = 0; nf < 4; ++nf) o[nf] = zero8();

  const half_t* Kbase = Kh + (size_t)bh * S_ * HD_;
  const half_t* Vbase = Vh + (size_t)bh * S_ * HD_;

#if defined(TDM_OK)
  const unsigned ldsKs = (unsigned)(size_t)(const void*)Ks;  // LDS byte address
#endif

  for (int kb = 0; kb < S_; kb += 32) {
    // ---- stage K block (TDM DMA if available, else vector copy) ----
#if defined(TDM_OK)
    if (w == 0) {
      const unsigned long long ga =
          (unsigned long long)(size_t)(const void*)(Kbase + (size_t)kb * HD_);
      u32x4 g0;
      g0[0] = 1u;                                            // count=1, user mode
      g0[1] = ldsKs;                                         // lds_addr
      g0[2] = (unsigned)ga;                                  // global_addr[31:0]
      g0[3] = (unsigned)((ga >> 32) & 0x01FFFFFFu) | 0x80000000u; // addr[56:32] | type=2
      i32x8 g1;
      g1[0] = 0x00010000;          // data_size = 1 (2 bytes)
      g1[1] = (int)(2048u << 16);  // tensor_dim0 = 2048 (lo16 at bits 63:48)
      g1[2] = (int)(1u << 16);     // tensor_dim0 hi16 = 0, tensor_dim1 = 1
      g1[3] = (int)(2048u << 16);  // tensor_dim1 hi16 = 0, tile_dim0 = 2048
      g1[4] = 0;                   // tile_dim1 = tile_dim2 = 0 (unused)
      g1[5] = 2048;                // tensor_dim0_stride lo32
      g1[6] = 0;
      g1[7] = 0;
      i32x4 gz; gz[0] = gz[1] = gz[2] = gz[3] = 0;
#if defined(TDM_SIX)
      i32x8 gz8;
#pragma unroll
      for (int i = 0; i < 8; ++i) gz8[i] = 0;
      __builtin_amdgcn_tensor_load_to_lds(g0, g1, gz, gz, gz8, 0);
#else
      __builtin_amdgcn_tensor_load_to_lds(g0, g1, gz, gz, 0);
#endif
      wait_tensorcnt0();
    }
#else
    *(v8h*)(Ks + t * 8) = *(const v8h*)(Kbase + (size_t)kb * HD_ + t * 8);
#endif

    // ---- stage V^T with packed pair stores (128 active threads) ----
    if (t < 128) {
      const int kk2 = t & 15;          // key pair (keys 2*kk2, 2*kk2+1)
      const int hd0 = (t >> 4) * 8;    // 8 hd values
      const half_t* vp0 = Vbase + (size_t)(kb + 2 * kk2) * HD_ + hd0;
      v8h a0 = *(const v8h*)vp0;
      v8h a1 = *(const v8h*)(vp0 + HD_);
#pragma unroll
      for (int i = 0; i < 8; ++i) {
        v2h pr; pr[0] = a0[i]; pr[1] = a1[i];
        *(v2h*)(VTs + (hd0 + i) * 32 + 2 * kk2) = pr;
      }
    }
    __syncthreads();

    if (kb + 32 < S_) {  // prefetch next V (and K on the manual path)
      __builtin_prefetch(Vbase + (size_t)(kb + 32) * HD_ + t * 8, 0, 0);
#if !defined(TDM_OK)
      __builtin_prefetch(Kbase + (size_t)(kb + 32) * HD_ + t * 8, 0, 0);
#endif
    }

    // ---- scores: 16 rows x 32 keys, scaled + cdr bias ----
    v8f sc[2];
#pragma unroll
    for (int n2 = 0; n2 < 2; ++n2) {
      const half_t* kp = &Ks[(n2 * 16 + nl) * 64];
      v16h b0 = cat16(*(const v8h*)(kp + 8 * hi), *(const v8h*)(kp + 16 + 8 * hi));
      v16h b1 = cat16(*(const v8h*)(kp + 32 + 8 * hi), *(const v8h*)(kp + 32 + 16 + 8 * hi));
      v8f z = zero8();
      z = wmma16(aq[0], b0, z);
      z = wmma16(aq[1], b1, z);
      const float biasv = cdr_bias[b * S_ + kb + n2 * 16 + nl] * cw;
#pragma unroll
      for (int r = 0; r < 8; ++r) z[r] = z[r] * 0.125f + biasv;
      sc[n2] = z;
    }

    // ---- online softmax per row (rows match C-layout halves) ----
#pragma unroll
    for (int r = 0; r < 8; ++r) {
      float tm = fmaxf(sc[0][r], sc[1][r]);
#pragma unroll
      for (int d = 1; d < 16; d <<= 1) tm = fmaxf(tm, __shfl_xor(tm, d, 32));
      const float nm = fmaxf(rm[r], tm);
      const float al = __expf(rm[r] - nm);
      const float p0 = __expf(sc[0][r] - nm);
      const float p1 = __expf(sc[1][r] - nm);
      float ps = p0 + p1;
#pragma unroll
      for (int d = 1; d < 16; d <<= 1) ps += __shfl_xor(ps, d, 32);
      rl[r] = rl[r] * al + ps;
      rm[r] = nm;
#pragma unroll
      for (int nf = 0; nf < 4; ++nf) o[nf][r] *= al;
      const int Mrow = r + 8 * hi;
      Ps[w][Mrow * 32 + nl]      = (half_t)p0;   // C-layout -> LDS
      Ps[w][Mrow * 32 + 16 + nl] = (half_t)p1;
    }

    // reload P in A-fragment layout (same-wave DS ops are in order)
    const half_t* pp = &Ps[w][nl * 32];
    v16h pa = cat16(*(const v8h*)(pp + 8 * hi), *(const v8h*)(pp + 16 + 8 * hi));

    // ---- O += P @ V ----
#pragma unroll
    for (int nf = 0; nf < 4; ++nf) {
      const half_t* vp = &VTs[(nf * 16 + nl) * 32];
      v16h bvv = cat16(*(const v8h*)(vp + 8 * hi), *(const v8h*)(vp + 16 + 8 * hi));
      o[nf] = wmma16(pa, bvv, o[nf]);
    }
    __syncthreads();
  }

  // normalize and write context in [B,S,D] (d = h*HD+hd) as f16
#pragma unroll
  for (int r = 0; r < 8; ++r) {
    const float inv = 1.0f / rl[r];
    const int m = qbase + w * 16 + r + 8 * hi;
    half_t* op = attnh + ((size_t)b * S_ + m) * D_ + h * HD_;
#pragma unroll
    for (int nf = 0; nf < 4; ++nf)
      op[nf * 16 + nl] = (half_t)(o[nf][r] * inv);
  }
}

// -----------------------------------------------------------------------------
// Kernel 3: out projection + bias + residual. Same tiling as kernel 1.
// -----------------------------------------------------------------------------
__global__ __launch_bounds__(256)
void outproj_kernel(const half_t* __restrict__ attnh, const float* __restrict__ Wo,
                    const float* __restrict__ bo, const float* __restrict__ x,
                    float* __restrict__ res)
{
  __shared__ __align__(16) half_t As[128 * 32];
  __shared__ __align__(16) half_t Bs[64 * 32];

  const int Nbase = blockIdx.x * 64;
  const int Mbase = blockIdx.y * 128;
  const int t = threadIdx.x;
  const int w = t >> 5, lane = t & 31;
  const int hi = lane >> 4, nl = lane & 15;
  const int wm = w >> 1, wn = w & 1;

  v8f acc[2][2];
#pragma unroll
  for (int mi = 0; mi < 2; ++mi)
#pragma unroll
    for (int ni = 0; ni < 2; ++ni) acc[mi][ni] = zero8();

  for (int kb = 0; kb < D_; kb += 32) {
    // A tile: straight f16 vector copy
#pragma unroll
    for (int i = 0; i < 2; ++i) {
      int e = t * 16 + i * 8;
      int row = e >> 5, col = e & 31;
      *(v8h*)(As + e) = *(const v8h*)(attnh + (size_t)(Mbase + row) * D_ + kb + col);
    }
    // B tile transposed, packed conversion
#pragma unroll
    for (int i = 0; i < 4; ++i) {
      int p = t + 256 * i;
      int n = p & 63, kk = (p >> 6) * 2;
      const float* wp = &Wo[(size_t)(kb + kk) * D_ + Nbase + n];
      store_pk(Bs + n * 32 + kk, wp[0], wp[D_]);
    }
    __syncthreads();

    if (kb + 32 < D_) {
      __builtin_prefetch(attnh + (size_t)(Mbase + (t >> 1)) * D_ + kb + 32, 0, 0);
      __builtin_prefetch(&Wo[(size_t)(kb + 32 + (t >> 6)) * D_ + Nbase + (t & 63)], 0, 0);
    }

    v16h a[2], b[2];
#pragma unroll
    for (int mi = 0; mi < 2; ++mi) {
      const half_t* ap = &As[(wm * 32 + mi * 16 + nl) * 32];
      a[mi] = cat16(*(const v8h*)(ap + 8 * hi), *(const v8h*)(ap + 16 + 8 * hi));
    }
#pragma unroll
    for (int ni = 0; ni < 2; ++ni) {
      const half_t* bp = &Bs[(wn * 32 + ni * 16 + nl) * 32];
      b[ni] = cat16(*(const v8h*)(bp + 8 * hi), *(const v8h*)(bp + 16 + 8 * hi));
    }
#pragma unroll
    for (int mi = 0; mi < 2; ++mi)
#pragma unroll
      for (int ni = 0; ni < 2; ++ni)
        acc[mi][ni] = wmma16(a[mi], b[ni], acc[mi][ni]);
    __syncthreads();
  }

#pragma unroll
  for (int ni = 0; ni < 2; ++ni) {
    const int n = Nbase + wn * 32 + ni * 16 + nl;
    const float bval = bo[n];
#pragma unroll
    for (int mi = 0; mi < 2; ++mi) {
#pragma unroll
      for (int r = 0; r < 8; ++r) {
        const int m = Mbase + wm * 32 + mi * 16 + r + 8 * hi;
        res[(size_t)m * D_ + n] = acc[mi][ni][r] + bval + x[(size_t)m * D_ + n];
      }
    }
  }
}

// -----------------------------------------------------------------------------
// Kernel 4: LayerNorm over D=1024. grid = M_, block = 256.
// -----------------------------------------------------------------------------
__global__ __launch_bounds__(256)
void layernorm_kernel(const float* __restrict__ res, const float* __restrict__ gamma,
                      const float* __restrict__ beta, float* __restrict__ out)
{
  __shared__ float sred[256];
  const int row = blockIdx.x, t = threadIdx.x;
  const float* rp = res + (size_t)row * D_;
  float v[4], s = 0.0f, s2 = 0.0f;
#pragma unroll
  for (int i = 0; i < 4; ++i) {
    v[i] = rp[t + 256 * i];
    s += v[i];
    s2 += v[i] * v[i];
  }
  sred[t] = s; __syncthreads();
  for (int off = 128; off > 0; off >>= 1) {
    if (t < off) sred[t] += sred[t + off];
    __syncthreads();
  }
  const float mu = sred[0] * (1.0f / D_);
  __syncthreads();
  sred[t] = s2; __syncthreads();
  for (int off = 128; off > 0; off >>= 1) {
    if (t < off) sred[t] += sred[t + off];
    __syncthreads();
  }
  const float var = sred[0] * (1.0f / D_) - mu * mu;
  const float rstd = rsqrtf(var + 1e-5f);
#pragma unroll
  for (int i = 0; i < 4; ++i) {
    const int c = t + 256 * i;
    out[(size_t)row * D_ + c] = (v[i] - mu) * rstd * gamma[c] + beta[c];
  }
}

// -----------------------------------------------------------------------------
extern "C" void kernel_launch(void* const* d_in, const int* in_sizes, int n_in,
                              void* d_out, int out_size, void* d_ws, size_t ws_size,
                              hipStream_t stream)
{
  (void)in_sizes; (void)n_in; (void)out_size; (void)ws_size;
  const float* x        = (const float*)d_in[0];
  const float* cdr_bias = (const float*)d_in[1];
  const float* Wq = (const float*)d_in[2];
  const float* bq = (const float*)d_in[3];
  const float* Wk = (const float*)d_in[4];
  const float* bk = (const float*)d_in[5];
  const float* Wv = (const float*)d_in[6];
  const float* bv = (const float*)d_in[7];
  const float* Wo = (const float*)d_in[8];
  const float* bo = (const float*)d_in[9];
  const float* gamma = (const float*)d_in[10];
  const float* beta  = (const float*)d_in[11];
  const float* cw    = (const float*)d_in[12];
  float* out = (float*)d_out;

  char* ws = (char*)d_ws;
  const size_t HB = (size_t)B_ * S_ * D_ * sizeof(half_t);  // 8 MB
  half_t* Qh  = (half_t*)(ws);
  half_t* Kh  = (half_t*)(ws + HB);
  half_t* Vh  = (half_t*)(ws + 2 * HB);
  half_t* Ah  = (half_t*)(ws + 3 * HB);
  float*  res = (float*)(ws + 4 * HB);                      // 16 MB

  dim3 blk(256);
  qkv_rope_kernel<<<dim3(D_ / 64, M_ / 128, 3), blk, 0, stream>>>(
      x, Wq, bq, Wk, bk, Wv, bv, Qh, Kh, Vh);
  attention_kernel<<<dim3(S_ / 128, B_ * H_), blk, 0, stream>>>(
      Qh, Kh, Vh, cdr_bias, cw, Ah);
  outproj_kernel<<<dim3(D_ / 64, M_ / 128), blk, 0, stream>>>(
      Ah, Wo, bo, x, res);
  layernorm_kernel<<<dim3(M_), blk, 0, stream>>>(res, gamma, beta, out);
}
